// PointerNetwork_68513318306096
// MI455X (gfx1250) — matrix operand order repaired and checked by reading, not compiled, not run on hardware
//
#include <hip/hip_runtime.h>
#include <hip/hip_bf16.h>

#define NB 32
#define LSEQ 2048
#define DD 1024        // D2
#define AT 512         // ATTN
#define TT 8
#define NEGINF (-1e30f)

typedef __attribute__((ext_vector_type(16))) __bf16 v16bf;
typedef __attribute__((ext_vector_type(8)))  float  v8f;
typedef __attribute__((ext_vector_type(2)))  float  v2f;

// --- CDNA5 async global->LDS copy (ASYNCcnt-tracked), per-lane 16B ----------
__device__ __forceinline__ void async_copy_b128(uint32_t lds_off, const float* gptr) {
  asm volatile("global_load_async_to_lds_b128 %0, %1, off"
               :: "v"(lds_off), "v"(gptr)
               : "memory");
}
__device__ __forceinline__ void wait_asynccnt0() {
  asm volatile("s_wait_asynccnt 0x0" ::: "memory");
}

// A fragment (16x32 bf16 slice of one row): lanes 0-15 K={0..7,16..23}, 16-31 +8
__device__ __forceinline__ v16bf load_afrag(const float* __restrict__ rowp, int kh) {
  float buf[16];
  *(float4*)(buf + 0)  = *(const float4*)(rowp + kh * 8);
  *(float4*)(buf + 4)  = *(const float4*)(rowp + kh * 8 + 4);
  *(float4*)(buf + 8)  = *(const float4*)(rowp + 16 + kh * 8);
  *(float4*)(buf + 12) = *(const float4*)(rowp + 16 + kh * 8 + 4);
  v16bf a;
#pragma unroll
  for (int i = 0; i < 16; ++i) a[i] = (__bf16)buf[i];
  return a;
}

// B fragment (32x16 bf16): lane = col; lanes 0-15 K={0..7,16..23}, 16-31 +8
__device__ __forceinline__ v16bf load_bfrag(const float* __restrict__ colp) {
  float buf[16];
  *(float4*)(buf + 0)  = *(const float4*)(colp + 0);
  *(float4*)(buf + 4)  = *(const float4*)(colp + 4);
  *(float4*)(buf + 8)  = *(const float4*)(colp + 16);
  *(float4*)(buf + 12) = *(const float4*)(colp + 20);
  v16bf b;
#pragma unroll
  for (int i = 0; i < 16; ++i) b[i] = (__bf16)buf[i];
  return b;
}

// ---------------------------------------------------------------------------
// enc_proj[n,l,a] = sum_d enc[n,l,d] * W2[a,d] + b2[a]   (stored bf16)
// M = NB*LSEQ = 65536, K = DD = 1024, N = AT = 512.
// Block (8 waves) = 32-row M-tile x all 512 cols; wave w = cols w*64..+63.
// A staged via async-to-LDS (double buffered, K-chunk 64); each B fragment
// feeds two WMMAs (rows 0-15 / 16-31). enc is read exactly once (268 MB).
// ---------------------------------------------------------------------------
#define KC 64
#define APAD 68   // padded row stride (floats) -> rows hit distinct LDS banks
#define MROWS 32

__global__ __launch_bounds__(256) void enc_proj_kernel(
    const float* __restrict__ enc, const float* __restrict__ W2,
    const float* __restrict__ b2, __bf16* __restrict__ ep)
{
  __shared__ float sA[2][MROWS * APAD];

  const int tid  = threadIdx.x;
  const int wave = tid >> 5;
  const int lane = tid & 31;
  const int ln16 = lane & 15;
  const int kh   = lane >> 4;                 // K-half select
  const int mtile  = blockIdx.x;              // 0..2047 (32 rows each)
  const int ngroup = wave;                    // 0..7, 64 cols each

  // staging: 256 threads x 8 floats = 32 rows x 64 K per chunk (2 x b128 each)
  const int sr = tid >> 3;                    // 0..31
  const int sc = (tid & 7) * 8;               // 0..56
  const float* __restrict__ gsrc = enc + ((size_t)mtile * MROWS + sr) * DD + sc;
  const uint32_t lds_d0a = (uint32_t)(uintptr_t)&sA[0][sr * APAD + sc];
  const uint32_t lds_d0b = lds_d0a + 16;
  const uint32_t lds_d1a = (uint32_t)(uintptr_t)&sA[1][sr * APAD + sc];
  const uint32_t lds_d1b = lds_d1a + 16;

  v8f acc[2][4] = {};

  async_copy_b128(lds_d0a, gsrc);             // chunk 0 -> buf 0
  async_copy_b128(lds_d0b, gsrc + 4);

  for (int kc = 0; kc < DD; kc += KC) {
    const int buf = (kc >> 6) & 1;
    wait_asynccnt0();
    __syncthreads();                          // chunk kc resident for all waves
    if (kc + KC < DD) {                       // prefetch next chunk -> other buf
      async_copy_b128(buf ? lds_d0a : lds_d1a, gsrc + kc + KC);
      async_copy_b128(buf ? lds_d0b : lds_d1b, gsrc + kc + KC + 4);
    }

    const float* __restrict__ Arow0 = &sA[buf][ln16 * APAD];
    const float* __restrict__ Arow1 = &sA[buf][(16 + ln16) * APAD];
#pragma unroll
    for (int kk = 0; kk < KC; kk += 32) {
      const v16bf a0 = load_afrag(Arow0 + kk, kh);
      const v16bf a1 = load_afrag(Arow1 + kk, kh);

      const float* __restrict__ bbase =
          W2 + (size_t)(ngroup * 64 + ln16) * DD + kc + kk + kh * 8;
      const v16bf b0 = load_bfrag(bbase + 0 * 16 * DD);
      const v16bf b1 = load_bfrag(bbase + 1 * 16 * DD);
      const v16bf b2f = load_bfrag(bbase + 2 * 16 * DD);
      const v16bf b3 = load_bfrag(bbase + 3 * 16 * DD);

      // 8 back-to-back WMMAs, all independent accumulators (no D->A/B hazards)
      acc[0][0] = __builtin_amdgcn_wmma_f32_16x16x32_bf16(false, a0, false, b0,  (short)0, acc[0][0], false, false);
      acc[1][0] = __builtin_amdgcn_wmma_f32_16x16x32_bf16(false, a1, false, b0,  (short)0, acc[1][0], false, false);
      acc[0][1] = __builtin_amdgcn_wmma_f32_16x16x32_bf16(false, a0, false, b1,  (short)0, acc[0][1], false, false);
      acc[1][1] = __builtin_amdgcn_wmma_f32_16x16x32_bf16(false, a1, false, b1,  (short)0, acc[1][1], false, false);
      acc[0][2] = __builtin_amdgcn_wmma_f32_16x16x32_bf16(false, a0, false, b2f, (short)0, acc[0][2], false, false);
      acc[1][2] = __builtin_amdgcn_wmma_f32_16x16x32_bf16(false, a1, false, b2f, (short)0, acc[1][2], false, false);
      acc[0][3] = __builtin_amdgcn_wmma_f32_16x16x32_bf16(false, a0, false, b3,  (short)0, acc[0][3], false, false);
      acc[1][3] = __builtin_amdgcn_wmma_f32_16x16x32_bf16(false, a1, false, b3,  (short)0, acc[1][3], false, false);
    }
    __syncthreads();                          // all reads of buf done before refill
  }

#pragma unroll
  for (int m = 0; m < 2; ++m) {
#pragma unroll
    for (int s = 0; s < 4; ++s) {
      const int col = ngroup * 64 + s * 16 + ln16;
      const float bias = b2[col];
#pragma unroll
      for (int v = 0; v < 8; ++v) {
        const size_t r = (size_t)mtile * MROWS + m * 16 + v + 8 * kh;
        ep[r * AT + col] = (__bf16)(acc[m][s][v] + bias);
      }
    }
  }
}

// ---------------------------------------------------------------------------
// Generic f32 GEMM via V_WMMA_F32_16X16X4_F32:  C[M,N] = A[M,K] @ W[N,K]^T (+bias)
// A addressed as A[row*lda + aoff + k] so x / h live in one [x|h] buffer.
// One wave per 16x16 C-tile; full f32 for the LSTM recurrence.
// ---------------------------------------------------------------------------
__global__ __launch_bounds__(128) void gemm_f32_wmma_kernel(
    const float* __restrict__ A, int lda, int aoff,
    const float* __restrict__ W, const float* __restrict__ bias,
    float* __restrict__ C, int K, int Nout)
{
  const int wave = threadIdx.x >> 5;
  const int lane = threadIdx.x & 31;
  const int ln16 = lane & 15;
  const int kh   = lane >> 4;
  const int gw = blockIdx.x * 4 + wave;
  const int ntiles = Nout >> 4;
  const int ntile = gw % ntiles;
  const int mtile = gw / ntiles;
  const int row = mtile * 16 + ln16;
  const int col = ntile * 16 + ln16;
  const float* __restrict__ ar = A + (size_t)row * lda + aoff;
  const float* __restrict__ wr = W + (size_t)col * K;

  v8f acc = {};
  for (int k0 = 0; k0 < K; k0 += 4) {
    const int ka = k0 + 2 * kh;          // lanes 0-15: K={k0,k0+1}; 16-31: {k0+2,k0+3}
    float2 av = *(const float2*)(ar + ka);
    float2 bv = *(const float2*)(wr + ka);
    v2f a, b;
    a[0] = av.x; a[1] = av.y;
    b[0] = bv.x; b[1] = bv.y;
    acc = __builtin_amdgcn_wmma_f32_16x16x4_f32(
        false, a, false, b, (short)0, acc, false, false);
  }
#pragma unroll
  for (int v = 0; v < 8; ++v) {
    const int r = mtile * 16 + v + 8 * kh;
    float outv = acc[v];
    if (bias) outv += bias[col];
    C[(size_t)r * Nout + col] = outv;
  }
}

// ---------------------------------------------------------------------------
// LSTM cell elementwise: gate order i,f,g,o. Writes new h into [x|h] buffer.
// ---------------------------------------------------------------------------
__global__ void lstm_cell_kernel(const float* __restrict__ gates,
    const float* __restrict__ b_ih, const float* __restrict__ b_hh,
    float* __restrict__ c, float* __restrict__ xh)
{
  const int tid = blockIdx.x * blockDim.x + threadIdx.x;   // NB*DD
  const int n = tid / DD, d = tid % DD;
  const size_t g0 = (size_t)n * 4 * DD;
  const float gi = gates[g0 + d]          + b_ih[d]          + b_hh[d];
  const float gf = gates[g0 + DD + d]     + b_ih[DD + d]     + b_hh[DD + d];
  const float gg = gates[g0 + 2 * DD + d] + b_ih[2 * DD + d] + b_hh[2 * DD + d];
  const float go = gates[g0 + 3 * DD + d] + b_ih[3 * DD + d] + b_hh[3 * DD + d];
  const float i = 1.f / (1.f + __expf(-gi));
  const float f = 1.f / (1.f + __expf(-gf));
  const float o = 1.f / (1.f + __expf(-go));
  const float g = tanhf(gg);
  const float cn = f * c[tid] + i * g;
  c[tid] = cn;
  xh[(size_t)n * (2 * DD) + DD + d] = o * tanhf(cn);
}

// ---------------------------------------------------------------------------
// attn[n,l] = w3 . tanh(q[n,:] + ep[n,l,:]) + b3 + mask_add   -> d_out[n,t,l]
// One wave per (n,l); wave32 xor-shuffle reduce.
// ---------------------------------------------------------------------------
__global__ __launch_bounds__(256) void attn_scores_kernel(
    const __bf16* __restrict__ ep, const float* __restrict__ q,
    const float* __restrict__ w3, const float* __restrict__ b3,
    const int* __restrict__ mask, float* __restrict__ out, int t)
{
  const int wave = threadIdx.x >> 5;
  const int lane = threadIdx.x & 31;
  const int rowid = blockIdx.x * 8 + wave;     // n*LSEQ + l
  const int n = rowid / LSEQ, l = rowid % LSEQ;
  const __bf16* __restrict__ er = ep + (size_t)rowid * AT;
  const float* __restrict__ qr = q + (size_t)n * AT;
  float s = 0.f;
#pragma unroll
  for (int i = 0; i < AT / 32; ++i) {
    const int a = lane + 32 * i;
    s += tanhf(qr[a] + (float)er[a]) * w3[a];
  }
#pragma unroll
  for (int off = 16; off > 0; off >>= 1) s += __shfl_xor(s, off, 32);
  if (lane == 0) {
    float val = s + b3[0];
    if (mask[(size_t)n * LSEQ + l] == 0) val += NEGINF;
    out[((size_t)n * TT + t) * LSEQ + l] = val;
  }
}

// ---------------------------------------------------------------------------
// Row softmax over L for batch row n (reads scores from d_out, writes p).
// ---------------------------------------------------------------------------
__global__ __launch_bounds__(256) void softmax_kernel(
    const float* __restrict__ out, float* __restrict__ p, int t)
{
  const int n = blockIdx.x;
  const int tid = threadIdx.x;
  const float* __restrict__ row = out + ((size_t)n * TT + t) * LSEQ;
  __shared__ float red[256];
  float m = -3.4e38f;
  for (int l = tid; l < LSEQ; l += 256) m = fmaxf(m, row[l]);
  red[tid] = m; __syncthreads();
  for (int s = 128; s > 0; s >>= 1) {
    if (tid < s) red[tid] = fmaxf(red[tid], red[tid + s]);
    __syncthreads();
  }
  m = red[0]; __syncthreads();
  float z = 0.f;
  for (int l = tid; l < LSEQ; l += 256) z += __expf(row[l] - m);
  red[tid] = z; __syncthreads();
  for (int s = 128; s > 0; s >>= 1) {
    if (tid < s) red[tid] += red[tid + s];
    __syncthreads();
  }
  const float inv = 1.f / red[0];
  for (int l = tid; l < LSEQ; l += 256)
    p[(size_t)n * LSEQ + l] = __expf(row[l] - m) * inv;
}

__global__ void zero_x_kernel(float* __restrict__ xh) {
  const int tid = blockIdx.x * blockDim.x + threadIdx.x;   // NB*DD
  const int n = tid / DD, d = tid % DD;
  xh[(size_t)n * (2 * DD) + d] = 0.f;
}

// ---------------------------------------------------------------------------
// x[n,d] += sum_l p[n,l]*enc[n,l,d]; L split into 16 chunks, f32 atomics.
// Pure HBM-bandwidth step; prefetch hides latency on the strided stream.
// ---------------------------------------------------------------------------
__global__ __launch_bounds__(256) void attend_kernel(
    const float* __restrict__ p, const float* __restrict__ enc,
    float* __restrict__ xh)
{
  const int n = blockIdx.x;
  const int d = blockIdx.y * 256 + threadIdx.x;
  const int l0 = blockIdx.z * (LSEQ / 16);
  const float* __restrict__ er = enc + ((size_t)n * LSEQ + l0) * DD + d;
  const float* __restrict__ pr = p + (size_t)n * LSEQ + l0;
  float s = 0.f;
  for (int l = 0; l < LSEQ / 16; ++l) {
    __builtin_prefetch(er + (size_t)(l + 4) * DD, 0, 1);
    s += pr[l] * er[(size_t)l * DD];
  }
  atomicAdd(&xh[(size_t)n * (2 * DD) + d], s);
}

__global__ void init_xh_kernel(const float* __restrict__ h0, float* __restrict__ xh) {
  const int tid = blockIdx.x * blockDim.x + threadIdx.x;   // NB*2*DD
  const int n = tid / (2 * DD), j = tid % (2 * DD);
  xh[tid] = (j < DD) ? 0.f : h0[(size_t)n * DD + (j - DD)];
}

__global__ void build_wcat_kernel(const float* __restrict__ Wih,
    const float* __restrict__ Whh, float* __restrict__ Wcat) {
  const size_t tid = (size_t)blockIdx.x * blockDim.x + threadIdx.x;  // 4*DD*2*DD
  const int j = (int)(tid / (2 * DD)), k = (int)(tid % (2 * DD));
  Wcat[tid] = (k < DD) ? Wih[(size_t)j * DD + k] : Whh[(size_t)j * DD + (k - DD)];
}

// ---------------------------------------------------------------------------
extern "C" void kernel_launch(void* const* d_in, const int* in_sizes, int n_in,
                              void* d_out, int out_size, void* d_ws, size_t ws_size,
                              hipStream_t stream) {
  const float* enc  = (const float*)d_in[0];
  const float* h0   = (const float*)d_in[1];
  const float* c0   = (const float*)d_in[2];
  const int*   mask = (const int*)d_in[3];
  const float* Wih  = (const float*)d_in[4];
  const float* Whh  = (const float*)d_in[5];
  const float* bih  = (const float*)d_in[6];
  const float* bhh  = (const float*)d_in[7];
  const float* W1   = (const float*)d_in[8];
  const float* b1   = (const float*)d_in[9];
  const float* W2   = (const float*)d_in[10];
  const float* b2   = (const float*)d_in[11];
  const float* w3   = (const float*)d_in[12];
  const float* b3   = (const float*)d_in[13];
  float* out = (float*)d_out;

  char* ws = (char*)d_ws;
  __bf16* ep = (__bf16*)ws;   ws += (size_t)NB * LSEQ * AT * 2;          // 64 MiB
  float* Wcat = (float*)ws;   ws += (size_t)4 * DD * 2 * DD * 4;         // 32 MiB
  float* xh   = (float*)ws;   ws += (size_t)NB * 2 * DD * 4;             // [x|h]
  float* c    = (float*)ws;   ws += (size_t)NB * DD * 4;
  float* gates= (float*)ws;   ws += (size_t)NB * 4 * DD * 4;
  float* q    = (float*)ws;   ws += (size_t)NB * AT * 4;
  float* p    = (float*)ws;   ws += (size_t)NB * LSEQ * 4;

  build_wcat_kernel<<<(4 * DD * 2 * DD) / 256, 256, 0, stream>>>(Wih, Whh, Wcat);
  init_xh_kernel<<<(NB * 2 * DD) / 256, 256, 0, stream>>>(h0, xh);
  hipMemcpyAsync(c, c0, (size_t)NB * DD * 4, hipMemcpyDeviceToDevice, stream);
  enc_proj_kernel<<<(NB * LSEQ) / MROWS, 256, 0, stream>>>(enc, W2, b2, ep);

  for (int t = 0; t < TT; ++t) {
    // gates = [x|h] @ [W_ih|W_hh]^T  : M=32, K=2048, N=4096  (512 waves)
    gemm_f32_wmma_kernel<<<(2 * (4 * DD / 16)) / 4, 128, 0, stream>>>(
        xh, 2 * DD, 0, Wcat, nullptr, gates, 2 * DD, 4 * DD);
    lstm_cell_kernel<<<(NB * DD) / 256, 256, 0, stream>>>(gates, bih, bhh, c, xh);
    // q = h @ W1^T + b1 : M=32, K=1024, N=512  (64 waves)
    gemm_f32_wmma_kernel<<<(2 * (AT / 16)) / 4, 128, 0, stream>>>(
        xh, 2 * DD, DD, W1, b1, q, DD, AT);
    attn_scores_kernel<<<(NB * LSEQ) / 8, 256, 0, stream>>>(ep, q, w3, b3, mask, out, t);
    if (t < TT - 1) {
      softmax_kernel<<<NB, 256, 0, stream>>>(out, p, t);
      zero_x_kernel<<<(NB * DD) / 256, 256, 0, stream>>>(xh);
      attend_kernel<<<dim3(NB, DD / 256, 16), 256, 0, stream>>>(p, enc, xh);
    }
  }
}
